// BahdanauAttention_27496380629512
// MI455X (gfx1250) — matrix operand order repaired and checked by reading, not compiled, not run on hardware
//
#include <hip/hip_runtime.h>
#include <hip/hip_bf16.h>

#define B_ 64
#define S_ 2048
#define H_ 512
#define TS 32            // sequence-tile rows per workgroup (scores kernel)
#define LDSTR 520        // padded LDS row stride in bf16 elems (bank-conflict free)

typedef __attribute__((ext_vector_type(16))) __bf16 v16bf;
typedef __attribute__((ext_vector_type(8)))  __bf16 v8bf;
typedef __attribute__((ext_vector_type(8)))  float  v8f;
typedef __attribute__((ext_vector_type(4)))  float  v4f;

// round-to-nearest-even fp32 -> bf16 (as 16-bit pattern in low bits)
static __device__ __forceinline__ unsigned bf16_rne(float x) {
    unsigned u = __float_as_uint(x);
    return (u + 0x7fffu + ((u >> 16) & 1u)) >> 16;
}

// branchless tanh: 1 - 2/(exp(2x)+1); exp2/rcp are single TRANS ops (co-execute with VALU)
static __device__ __forceinline__ float fast_tanh(float x) {
    float e = __builtin_amdgcn_exp2f(x * 2.885390081777927f);   // exp(2x)
    return 1.0f - 2.0f * __builtin_amdgcn_rcpf(e + 1.0f);
}

// ---------------- W_k fp32 -> bf16 (L2-resident, reused by all score blocks) ----
__global__ __launch_bounds__(256) void k_cvt_bf16(const float* __restrict__ src,
                                                  __bf16* __restrict__ dst, int n) {
    int i = blockIdx.x * 256 + threadIdx.x;
    if (i < n) dst[i] = (__bf16)src[i];
}

// ---------------- query = dh @ Wq^T  [B,H], tiny fp32 GEMM ----------------------
__global__ __launch_bounds__(256) void k_query(const float* __restrict__ dh,
                                               const float* __restrict__ Wq,
                                               float* __restrict__ q) {
    __shared__ float sdh[H_];
    const int b = blockIdx.x;
    for (int i = threadIdx.x; i < H_; i += 256) sdh[i] = dh[b * H_ + i];
    __syncthreads();
    for (int g = threadIdx.x; g < H_; g += 256) {
        const float* wr = Wq + (size_t)g * H_;
        float acc = 0.f;
        #pragma unroll 8
        for (int h = 0; h < H_; ++h) acc += sdh[h] * wr[h];
        q[b * H_ + g] = acc;
    }
}

// ---------------- scores: keys GEMM (bf16 WMMA) + tanh + dot(v), fused ----------
// grid = (S/TS, B), block = 256 (8 wave32). Wave w owns g in [64w, 64w+64).
__global__ __launch_bounds__(256) void k_scores(const float*  __restrict__ enc,   // [B,S,H] fp32
                                                const __bf16* __restrict__ Wk,    // [H,H] bf16
                                                const float*  __restrict__ q,     // [B,H]
                                                const float*  __restrict__ v,     // [H]
                                                float*        __restrict__ scores)// [B,S]
{
    __shared__ __bf16 lds_enc[TS * LDSTR];
    __shared__ float  lds_sc[TS];

    const int b   = blockIdx.y;
    const int s0  = blockIdx.x * TS;
    const int tid = threadIdx.x;

    // Stage enc tile [TS x H] fp32 -> bf16 into LDS.
    // Coalesced b128 global reads; packed RNE bf16 -> one ds_store_b64 per float4.
    const float* gsrc = enc + ((size_t)b * S_ + s0) * H_;
    #pragma unroll
    for (int j = 0; j < (TS * H_ / 4) / 256; ++j) {           // 16 iters
        int idx4 = tid + j * 256;
        int row  = idx4 >> 7;          // H_/4 = 128 float4 per row
        int c4   = idx4 & 127;
        v4f f = *(const v4f*)(gsrc + (size_t)row * H_ + c4 * 4);
        unsigned lo = bf16_rne(f[0]) | (bf16_rne(f[1]) << 16);
        unsigned hi = bf16_rne(f[2]) | (bf16_rne(f[3]) << 16);
        unsigned long long p = ((unsigned long long)hi << 32) | lo;
        *(unsigned long long*)&lds_enc[row * LDSTR + c4 * 4] = p;   // 8B-aligned (1040*row + 8*c4)
    }
    if (tid < TS) lds_sc[tid] = 0.f;
    __syncthreads();

    const int wave = tid >> 5;          // 0..7
    const int lane = tid & 31;
    const int half = lane >> 4;         // 0: low 16 lanes, 1: high 16 lanes
    const int l16  = lane & 15;
    const int g0   = wave * 64;

    // Per-lane q[g] and v[g] for each N-tile (g = g0 + ni*16 + l16).
    float qv[4], vv[4];
    #pragma unroll
    for (int ni = 0; ni < 4; ++ni) {
        int g = g0 + ni * 16 + l16;
        qv[ni] = q[b * H_ + g];
        vv[ni] = v[g];
    }

    // Fold the query add into the WMMA C-matrix: D layout has N = lane, so every
    // accumulator element of this lane shares the same g -> init all 8 to q[g].
    v8f acc[2][4];
    #pragma unroll
    for (int mi = 0; mi < 2; ++mi)
        #pragma unroll
        for (int ni = 0; ni < 4; ++ni) {
            v8f c0;
            #pragma unroll
            for (int r = 0; r < 8; ++r) c0[r] = qv[ni];
            acc[mi][ni] = c0;
        }

    // K loop over H in steps of 32 (bf16 WMMA depth).
    for (int kc = 0; kc < H_; kc += 32) {
        // A fragments: contiguous K order matching B — low lanes K 0-15, high lanes K 16-31.
        v16bf a[2];
        #pragma unroll
        for (int mi = 0; mi < 2; ++mi) {
            const __bf16* rp = &lds_enc[(mi * 16 + l16) * LDSTR + kc];
            v8bf lo = *(const v8bf*)(rp + half * 16);
            v8bf hi = *(const v8bf*)(rp + half * 16 + 8);
            #pragma unroll
            for (int e = 0; e < 8; ++e) { a[mi][e] = lo[e]; a[mi][8 + e] = hi[e]; }
        }
        // B fragments: B(k,n) = Wk[n][k]; lane holds column n = ni*16+l16,
        // 16 contiguous K values (half selects K 0-15 vs 16-31). 32B-aligned loads from L2.
        #pragma unroll
        for (int ni = 0; ni < 4; ++ni) {
            int g = g0 + ni * 16 + l16;
            v16bf bfrag = *(const v16bf*)(Wk + (size_t)g * H_ + kc + half * 16);
            #pragma unroll
            for (int mi = 0; mi < 2; ++mi)
                acc[mi][ni] = __builtin_amdgcn_wmma_f32_16x16x32_bf16(
                    false, a[mi], false, bfrag, (short)0, acc[mi][ni], false, false);
        }
    }

    // Epilogue: score[s] += tanh(q[g] + key[s,g]) * v[g]  (q already folded into acc).
    // C/D layout: VGPR r -> M = 8*half + r (within 16-row tile), lane -> N.
    float srow[2][8];
    #pragma unroll
    for (int mi = 0; mi < 2; ++mi)
        #pragma unroll
        for (int r = 0; r < 8; ++r) srow[mi][r] = 0.f;

    #pragma unroll
    for (int mi = 0; mi < 2; ++mi)
        #pragma unroll
        for (int ni = 0; ni < 4; ++ni)
            #pragma unroll
            for (int r = 0; r < 8; ++r)
                srow[mi][r] += fast_tanh(acc[mi][ni][r]) * vv[ni];

    #pragma unroll
    for (int mi = 0; mi < 2; ++mi)
        #pragma unroll
        for (int r = 0; r < 8; ++r)
            atomicAdd(&lds_sc[mi * 16 + 8 * half + r], srow[mi][r]);   // ds_add_f32
    __syncthreads();

    if (tid < TS) scores[(size_t)b * S_ + s0 + tid] = lds_sc[tid];
}

// ---------------- softmax over S per batch row --------------------------------
__global__ __launch_bounds__(256) void k_softmax(const float* __restrict__ scores,
                                                 float* __restrict__ attn) {
    __shared__ float red[256];
    const int b = blockIdx.x, tid = threadIdx.x;
    const float* row = scores + (size_t)b * S_;

    float m = -1e30f;
    for (int s = tid; s < S_; s += 256) m = fmaxf(m, row[s]);
    red[tid] = m; __syncthreads();
    for (int o = 128; o > 0; o >>= 1) { if (tid < o) red[tid] = fmaxf(red[tid], red[tid + o]); __syncthreads(); }
    m = red[0]; __syncthreads();

    float sum = 0.f;
    for (int s = tid; s < S_; s += 256) sum += __expf(row[s] - m);
    red[tid] = sum; __syncthreads();
    for (int o = 128; o > 0; o >>= 1) { if (tid < o) red[tid] += red[tid + o]; __syncthreads(); }
    const float inv = 1.0f / red[0];

    float* arow = attn + (size_t)b * S_;
    for (int s = tid; s < S_; s += 256) arow[s] = __expf(row[s] - m) * inv;
}

// ---------------- context = attn @ enc  (second streaming pass over enc) ------
// grid = (H/64, B), block = 256: 4 s-subgroups x 64 h columns (coalesced on h).
__global__ __launch_bounds__(256) void k_context(const float* __restrict__ enc,
                                                 const float* __restrict__ attn,
                                                 float* __restrict__ ctx) {
    __shared__ float red[4][64];
    const int b  = blockIdx.y;
    const int h0 = blockIdx.x * 64;
    const int tid = threadIdx.x;
    const int sg  = tid >> 6;     // 0..3
    const int hc  = tid & 63;

    const float* arow = attn + (size_t)b * S_;
    const float* e    = enc + (size_t)b * S_ * H_ + h0 + hc;
    float acc = 0.f;
    for (int s = sg; s < S_; s += 4)
        acc += arow[s] * e[(size_t)s * H_];
    red[sg][hc] = acc;
    __syncthreads();
    if (sg == 0)
        ctx[(size_t)b * H_ + h0 + hc] = red[0][hc] + red[1][hc] + red[2][hc] + red[3][hc];
}

extern "C" void kernel_launch(void* const* d_in, const int* in_sizes, int n_in,
                              void* d_out, int out_size, void* d_ws, size_t ws_size,
                              hipStream_t stream) {
    const float* dh  = (const float*)d_in[0];   // [B,H]
    const float* enc = (const float*)d_in[1];   // [B,S,H]
    const float* Wq  = (const float*)d_in[2];   // [H,H]
    const float* Wk  = (const float*)d_in[3];   // [H,H]
    const float* v   = (const float*)d_in[4];   // [H]

    float* out  = (float*)d_out;
    float* ctx  = out;                 // [B,H]   (return order: context first)
    float* attn = out + B_ * H_;       // [B,S]

    // Workspace: query [B*H] f32 | scores [B*S] f32 | Wk bf16 [H*H]  (~1.2 MB)
    float*  q      = (float*)d_ws;
    float*  scores = q + B_ * H_;
    __bf16* Wkb    = (__bf16*)(scores + (size_t)B_ * S_);

    k_cvt_bf16<<<(H_ * H_ + 255) / 256, 256, 0, stream>>>(Wk, Wkb, H_ * H_);
    k_query<<<B_, 256, 0, stream>>>(dh, Wq, q);

    dim3 gS(S_ / TS, B_);
    k_scores<<<gS, 256, 0, stream>>>(enc, Wkb, q, v, scores);

    k_softmax<<<B_, 256, 0, stream>>>(scores, attn);

    dim3 gC(H_ / 64, B_);
    k_context<<<gC, 256, 0, stream>>>(enc, attn, ctx);
}
